// LPFA_8521215115538
// MI455X (gfx1250) — matrix operand order repaired and checked by reading, not compiled, not run on hardware
//
#include <hip/hip_runtime.h>
#include <hip/hip_bf16.h>

// ---------------- problem constants ----------------
#define B_ 8
#define C_ 64
#define N_ 4096
#define K_ 20
#define O_ 64
constexpr int   CNT   = B_ * N_ * K_;   // 655360 samples
constexpr float EPS_  = 1e-5f;
constexpr float SLOPE = 0.2f;

// ---------------- workspace layout (float offsets) ----------------
constexpr int OFF_PFS  = 0;     // 9   : sum of pf
constexpr int OFF_PFQ  = 16;    // 45  : upper-tri second moments of pf
constexpr int OFF_SUM1 = 64;    // 64  : sum  of t1 per channel
constexpr int OFF_SQ1  = 128;   // 64  : sum2 of t1
constexpr int OFF_SUM2 = 192;   // 64
constexpr int OFF_SQ2  = 256;   // 64
constexpr int OFF_STATS_BYTES = 320 * 4;   // memset region
constexpr int OFF_SC0 = 320, OFF_BI0 = 384;
constexpr int OFF_SC1 = 448, OFF_BI1 = 512;
constexpr int OFF_SC2 = 576, OFF_BI2 = 640;
constexpr int OFF_W1BF = 704;   // 4096 bf16 = 2048 floats
constexpr int OFF_W2BF = 2752;  // 4096 bf16
// total ws use: 4800 floats = 19.2 KB

// ---------------- types ----------------
typedef __bf16 bf16;
typedef __attribute__((ext_vector_type(16))) __bf16        v16bf;
typedef __attribute__((ext_vector_type(8)))  float         v8f;
typedef __attribute__((ext_vector_type(4)))  unsigned int  u4;

__device__ __forceinline__ float lrelu(float z) { return z >= 0.f ? z : SLOPE * z; }

__device__ __forceinline__ unsigned int pk2(float a, float b) {
    union { bf16 h[2]; unsigned int u; } t;
    t.h[0] = (bf16)a; t.h[1] = (bf16)b;
    return t.u;
}
__device__ __forceinline__ u4 mk4(unsigned a, unsigned b, unsigned c, unsigned d) {
    u4 v; v.x = a; v.y = b; v.z = c; v.w = d; return v;
}
// A-fragment: two 16B chunks at element offsets base and base+8 (contiguous k order used here)
__device__ __forceinline__ v16bf load_frag_g(const bf16* p) {
    union { v16bf v; u4 q[2]; } t;
    const u4* s = (const u4*)p;
    t.q[0] = s[0];
    t.q[1] = s[1];          // +8 bf16 = +16B = +1 u4
    return t.v;
}
// B-fragment: 32 contiguous bytes in LDS (two consecutive u4)
__device__ __forceinline__ v16bf load_frag_l(const u4* p) {
    union { v16bf v; u4 q[2]; } t;
    t.q[0] = p[0]; t.q[1] = p[1];
    return t.v;
}
__device__ __forceinline__ v8f wmma_bf16(v16bf a, v16bf b, v8f c) {
    return __builtin_amdgcn_wmma_f32_16x16x32_bf16(false, a, false, b, (short)0, c, false, false);
}

// ---------------- weight fp32 -> bf16 ----------------
__global__ void wconv_kernel(const float* __restrict__ W1, const float* __restrict__ W2,
                             float* __restrict__ ws) {
    int i = blockIdx.x * 256 + threadIdx.x;
    if (i < 4096) {
        bf16* w1b = (bf16*)(ws + OFF_W1BF);
        bf16* w2b = (bf16*)(ws + OFF_W2BF);
        w1b[i] = (bf16)W1[i];
        w2b[i] = (bf16)W2[i];
    }
}

// ---------------- pf moments (for layer0 BN stats, exact via linearity) ----------------
__global__ void pfmom_kernel(const float* __restrict__ xyz, const int* __restrict__ idx,
                             float* __restrict__ ws) {
    __shared__ float sS[9];
    __shared__ float sQ[45];
    int tid = threadIdx.x;
    if (tid < 9)  sS[tid] = 0.f;
    if (tid < 45) sQ[tid] = 0.f;
    __syncthreads();

    int e = blockIdx.x * 256 + tid;
    if (e < CNT) {
        int b = e / (N_ * K_);
        int r = e - b * (N_ * K_);
        int n = r / K_;
        int k = r - n * K_;
        int j = idx[(b * N_ + n) * K_ + k];
        const float* zb = xyz + (size_t)b * 3 * N_;
        float pf[9];
        pf[0] = zb[n]; pf[1] = zb[N_ + n]; pf[2] = zb[2 * N_ + n];
        pf[3] = zb[j]; pf[4] = zb[N_ + j]; pf[5] = zb[2 * N_ + j];
        pf[6] = pf[3] - pf[0]; pf[7] = pf[4] - pf[1]; pf[8] = pf[5] - pf[2];
        #pragma unroll
        for (int d = 0; d < 9; ++d) atomicAdd(&sS[d], pf[d]);
        int qi = 0;
        #pragma unroll
        for (int d = 0; d < 9; ++d)
            #pragma unroll
            for (int f = d; f < 9; ++f) { atomicAdd(&sQ[qi], pf[d] * pf[f]); ++qi; }
    }
    __syncthreads();
    if (tid < 9)  atomicAdd(&ws[OFF_PFS + tid], sS[tid]);
    if (tid < 45) atomicAdd(&ws[OFF_PFQ + tid], sQ[tid]);
}

// ---------------- derive layer0 scale/bias:  BN(W0*pf) stats from pf moments ----------------
__global__ void stats0_kernel(float* __restrict__ ws, const float* __restrict__ W0,
                              const float* __restrict__ g0, const float* __restrict__ b0) {
    int o = threadIdx.x;           // 64 threads
    const float inv = 1.0f / (float)CNT;
    float wrow[9];
    #pragma unroll
    for (int d = 0; d < 9; ++d) wrow[d] = W0[o * 9 + d];
    float mu = 0.f;
    #pragma unroll
    for (int d = 0; d < 9; ++d) mu += wrow[d] * (ws[OFF_PFS + d] * inv);
    float ex2 = 0.f; int qi = 0;
    for (int d = 0; d < 9; ++d)
        for (int f = d; f < 9; ++f) {
            float q  = ws[OFF_PFQ + qi] * inv; ++qi;
            float wp = wrow[d] * wrow[f];
            ex2 += (d == f ? wp : 2.f * wp) * q;
        }
    float var = ex2 - mu * mu;
    float sc  = g0[o] * rsqrtf(var + EPS_);
    ws[OFF_SC0 + o] = sc;
    ws[OFF_BI0 + o] = b0[o] - mu * sc;
}

// ---------------- derive layer1/2 scale/bias from accumulated sum/sumsq ----------------
__global__ void statsL_kernel(float* __restrict__ ws, int offS, int offQ,
                              const float* __restrict__ g, const float* __restrict__ bb,
                              int offSc, int offBi) {
    int o = threadIdx.x;
    const float inv = 1.0f / (float)CNT;
    float mean = ws[offS + o] * inv;
    float var  = ws[offQ + o] * inv - mean * mean;
    float sc   = g[o] * rsqrtf(var + EPS_);
    ws[offSc + o] = sc;
    ws[offBi + o] = bb[o] - mean * sc;
}

// ---------------- main fused pass ----------------
// PHASE 1: layer0 + layer1 WMMA, accumulate t1 stats
// PHASE 2: + BN1/lrelu + layer2 WMMA, accumulate t2 stats
// PHASE 3: + BN2/lrelu + mean over K -> out
constexpr int HU4 = 9;   // u4 per h-row (72 bf16 = 144B, 16B aligned, conflict-free)
constexpr int H2S = 68;  // float stride for phase-3 staging

template <int PHASE>
__global__ __launch_bounds__(64, 4) void pass_kernel(const float* __restrict__ x,
                                                     const float* __restrict__ xyz,
                                                     const int*   __restrict__ idx,
                                                     const float* __restrict__ W0,
                                                     float*       __restrict__ ws,
                                                     float*       __restrict__ out) {
    __shared__ float sW0[64 * 9];
    __shared__ float sSc0[64], sBi0[64];
    __shared__ float sSc1[64], sBi1[64];
    __shared__ float sSc2[64], sBi2[64];
    __shared__ u4    hbuf[2 * 16 * HU4];                     // per-wave bf16 h staging
    __shared__ float sAcc[128];
    __shared__ float h2buf[(PHASE == 3) ? 2 * 80 * H2S : 2]; // per-wave fp32 h2 staging

    const int tid = threadIdx.x;
    for (int i = tid; i < 576; i += 64) sW0[i] = W0[i];
    if (tid < 64) {
        sSc0[tid] = ws[OFF_SC0 + tid]; sBi0[tid] = ws[OFF_BI0 + tid];
        if (PHASE >= 2) { sSc1[tid] = ws[OFF_SC1 + tid]; sBi1[tid] = ws[OFF_BI1 + tid]; }
        if (PHASE == 3) { sSc2[tid] = ws[OFF_SC2 + tid]; sBi2[tid] = ws[OFF_BI2 + tid]; }
    }
    if (PHASE < 3) { if (tid < 64) { sAcc[tid] = 0.f; sAcc[64 + tid] = 0.f; } }
    __syncthreads();

    const int lane = tid & 31;
    const int wv   = tid >> 5;
    const int kh   = lane >> 4;   // lane half
    const int m    = lane & 15;   // row within tile

    // ---- W1 A-fragments kept resident (contiguous k order: lane m = M-row, kh = K half) ----
    // W2 fragments are re-loaded from L2/WGP$ each tile to stay under 256 VGPRs.
    const bf16* w1b = (const bf16*)(ws + OFF_W1BF);
    const bf16* w2b = (const bf16*)(ws + OFF_W2BF);
    v16bf a1[4][2];
    #pragma unroll
    for (int ot = 0; ot < 4; ++ot)
        #pragma unroll
        for (int kc = 0; kc < 2; ++kc)
            a1[ot][kc] = load_frag_g(w1b + (ot * 16 + m) * 64 + kc * 32 + kh * 16);

    // ---- wave task: (batch b, 4 consecutive points) = 80 samples = 5 WMMA tiles ----
    const int task = blockIdx.x * 2 + wv;       // 8192 tasks
    const int b    = task >> 10;
    const int n0   = (task & 1023) << 2;
    const float* xb = x   + (size_t)b * C_ * N_;
    const float* zb = xyz + (size_t)b * 3 * N_;
    const int*   ib = idx + (size_t)b * N_ * K_;

    u4*    hw  = hbuf + wv * 16 * HU4;
    float* h2w = h2buf + ((PHASE == 3) ? wv * 80 * H2S : 0);

    float accS[32], accQ[32];
    #pragma unroll
    for (int i = 0; i < 32; ++i) { accS[i] = 0.f; accQ[i] = 0.f; }

    for (int mt = 0; mt < 5; ++mt) {
        const int s  = mt * 16 + m;          // 0..79
        const int p  = s / 20;
        const int kk = s - p * 20;
        const int n  = n0 + p;
        const int j  = ib[n * K_ + kk];

        float pf[9];
        pf[0] = zb[n]; pf[1] = zb[N_ + n]; pf[2] = zb[2 * N_ + n];
        pf[3] = zb[j]; pf[4] = zb[N_ + j]; pf[5] = zb[2 * N_ + j];
        pf[6] = pf[3] - pf[0]; pf[7] = pf[4] - pf[1]; pf[8] = pf[5] - pf[2];

        // ---- layer0 (fp32) + feat gather + BN0 + lrelu; 4 groups of 8 channels ----
        // grouped + unroll 1 to keep live VGPRs low (occupancy hides gather latency)
        const int cb = kh * 32;
        u4* dstrow = hw + m * HU4 + kh * 4;
        #pragma unroll 1
        for (int g = 0; g < 4; ++g) {
            unsigned uu[4];
            #pragma unroll
            for (int h2i = 0; h2i < 4; ++h2i) {
                float v01[2];
                #pragma unroll
                for (int q = 0; q < 2; ++q) {
                    int c = cb + g * 8 + h2i * 2 + q;
                    const float* wr = &sW0[c * 9];
                    float t = 0.f;
                    #pragma unroll
                    for (int d = 0; d < 9; ++d) t += wr[d] * pf[d];
                    float f = xb[c * N_ + j] - xb[c * N_ + n];   // gather (L2-resident)
                    v01[q] = lrelu(t * sSc0[c] + sBi0[c] + f);
                }
                uu[h2i] = pk2(v01[0], v01[1]);
            }
            dstrow[g] = mk4(uu[0], uu[1], uu[2], uu[3]);
        }

        // ---- layer1: B-fragments (contiguous k order: lane%16 = N, lane half = K half) ----
        const u4* hr = hw + m * HU4;
        v16bf bA = load_frag_l(hr + 2 * kh);        // K(c) 0..31
        v16bf bB = load_frag_l(hr + 4 + 2 * kh);    // K(c) 32..63
        v8f d[4];
        #pragma unroll
        for (int ot = 0; ot < 4; ++ot) {
            v8f acc = {};
            acc = wmma_bf16(a1[ot][0], bA, acc);
            acc = wmma_bf16(a1[ot][1], bB, acc);
            d[ot] = acc;
        }

        if (PHASE == 1) {
            #pragma unroll
            for (int ot = 0; ot < 4; ++ot)
                #pragma unroll
                for (int r = 0; r < 8; ++r) {
                    float v = d[ot][r];
                    accS[ot * 8 + r] += v;
                    accQ[ot * 8 + r] += v * v;
                }
        } else {
            // ---- BN1 + lrelu -> bf16 h1 -> LDS (overwrite h, same-wave DS is in-order) ----
            #pragma unroll
            for (int ot = 0; ot < 4; ++ot) {
                const int o0 = ot * 16 + kh * 8;
                float w[8];
                #pragma unroll
                for (int r = 0; r < 8; ++r) w[r] = lrelu(d[ot][r] * sSc1[o0 + r] + sBi1[o0 + r]);
                hw[m * HU4 + ot * 2 + kh] =
                    mk4(pk2(w[0], w[1]), pk2(w[2], w[3]), pk2(w[4], w[5]), pk2(w[6], w[7]));
            }
            v16bf cA2 = load_frag_l(hr + 2 * kh);
            v16bf cB2 = load_frag_l(hr + 4 + 2 * kh);
            v8f e2v[4];
            #pragma unroll
            for (int ot = 0; ot < 4; ++ot) {
                // reload W2 A-fragments (hits WGP$/L2; frees 64 VGPRs vs resident copy)
                const bf16* wbase = w2b + (ot * 16 + m) * 64 + kh * 16;
                v16bf a20 = load_frag_g(wbase);
                v16bf a21 = load_frag_g(wbase + 32);
                v8f acc = {};
                acc = wmma_bf16(a20, cA2, acc);
                acc = wmma_bf16(a21, cB2, acc);
                e2v[ot] = acc;
            }
            if (PHASE == 2) {
                #pragma unroll
                for (int ot = 0; ot < 4; ++ot)
                    #pragma unroll
                    for (int r = 0; r < 8; ++r) {
                        float v = e2v[ot][r];
                        accS[ot * 8 + r] += v;
                        accQ[ot * 8 + r] += v * v;
                    }
            } else {  // PHASE 3: BN2 + lrelu, pre-scale by 1/K, stage for K-mean
                #pragma unroll
                for (int ot = 0; ot < 4; ++ot)
                    #pragma unroll
                    for (int r = 0; r < 8; ++r) {
                        int o = ot * 16 + kh * 8 + r;
                        float v = lrelu(e2v[ot][r] * sSc2[o] + sBi2[o]) * (1.0f / (float)K_);
                        h2w[s * H2S + o] = v;
                    }
            }
        }
    }

    if (PHASE < 3) {
        #pragma unroll
        for (int ot = 0; ot < 4; ++ot)
            #pragma unroll
            for (int r = 0; r < 8; ++r) {
                int o = ot * 16 + kh * 8 + r;
                atomicAdd(&sAcc[o],      accS[ot * 8 + r]);
                atomicAdd(&sAcc[64 + o], accQ[ot * 8 + r]);
            }
        __syncthreads();
        const int offS = (PHASE == 1) ? OFF_SUM1 : OFF_SUM2;
        const int offQ = (PHASE == 1) ? OFF_SQ1  : OFF_SQ2;
        if (tid < 64) {
            atomicAdd(&ws[offS + tid], sAcc[tid]);
            atomicAdd(&ws[offQ + tid], sAcc[64 + tid]);
        }
    } else {
        __syncthreads();
        // wave-local K-mean: 4 points x 64 channels = 256 outputs per wave
        #pragma unroll
        for (int i2 = 0; i2 < 8; ++i2) {
            int u = lane + 32 * i2;
            int p = u >> 6;
            int o = u & 63;
            float sum = 0.f;
            #pragma unroll
            for (int k = 0; k < 20; ++k) sum += h2w[(p * 20 + k) * H2S + o];
            out[((size_t)(b * O_ + o)) * N_ + n0 + p] = sum;
        }
    }
}

// ---------------- host-side launch ----------------
extern "C" void kernel_launch(void* const* d_in, const int* in_sizes, int n_in,
                              void* d_out, int out_size, void* d_ws, size_t ws_size,
                              hipStream_t stream) {
    const float* x   = (const float*)d_in[0];
    const float* xyz = (const float*)d_in[1];
    const int*   idx = (const int*)  d_in[2];
    const float* W0  = (const float*)d_in[3];
    const float* g0  = (const float*)d_in[4];
    const float* b0  = (const float*)d_in[5];
    const float* W1  = (const float*)d_in[6];
    const float* g1  = (const float*)d_in[7];
    const float* b1  = (const float*)d_in[8];
    const float* W2  = (const float*)d_in[9];
    const float* g2  = (const float*)d_in[10];
    const float* b2  = (const float*)d_in[11];
    float* out = (float*)d_out;
    float* ws  = (float*)d_ws;
    (void)in_sizes; (void)n_in; (void)out_size; (void)ws_size;

    // zero the stat accumulators (graph-capture safe)
    hipMemsetAsync(ws, 0, OFF_STATS_BYTES, stream);

    wconv_kernel<<<16, 256, 0, stream>>>(W1, W2, ws);
    pfmom_kernel<<<(CNT + 255) / 256, 256, 0, stream>>>(xyz, idx, ws);
    stats0_kernel<<<1, 64, 0, stream>>>(ws, W0, g0, b0);

    pass_kernel<1><<<4096, 64, 0, stream>>>(x, xyz, idx, W0, ws, nullptr);
    statsL_kernel<<<1, 64, 0, stream>>>(ws, OFF_SUM1, OFF_SQ1, g1, b1, OFF_SC1, OFF_BI1);

    pass_kernel<2><<<4096, 64, 0, stream>>>(x, xyz, idx, W0, ws, nullptr);
    statsL_kernel<<<1, 64, 0, stream>>>(ws, OFF_SUM2, OFF_SQ2, g2, b2, OFF_SC2, OFF_BI2);

    pass_kernel<3><<<4096, 64, 0, stream>>>(x, xyz, idx, W0, ws, out);
}
